// ConditionalDetrTransformer_56968446214257
// MI455X (gfx1250) — matrix operand order repaired and checked by reading, not compiled, not run on hardware
//
#include <hip/hip_runtime.h>
#include <hip/hip_bf16.h>

#define BS   64
#define NQ   3000
#define NC   256
#define FH   100
#define FW   100
#define TOPK 1500
#define KPAD 1504          // TOPK rounded up to multiple of 32

typedef _Float16 v16h __attribute__((ext_vector_type(16)));
typedef float    v8f  __attribute__((ext_vector_type(8)));

// ---------------------------------------------------------------------------
// Kernel 1: scores[b,q] = max over 256 classes. One wave per row, float4 loads.
// Dominant memory phase: 197 MB streamed from HBM (~8.4 us at 23.3 TB/s).
// ---------------------------------------------------------------------------
__global__ __launch_bounds__(256) void scores_kernel(const float* __restrict__ cls,
                                                     float* __restrict__ scores) {
    int wid  = threadIdx.x >> 5;
    int lane = threadIdx.x & 31;
    int row  = blockIdx.x * 8 + wid;
    if (row >= BS * NQ) return;
    const float4* p = (const float4*)(cls + (size_t)row * NC + lane * 8);
    float4 x0 = p[0];
    float4 x1 = p[1];
    float m = fmaxf(fmaxf(fmaxf(x0.x, x0.y), fmaxf(x0.z, x0.w)),
                    fmaxf(fmaxf(x1.x, x1.y), fmaxf(x1.z, x1.w)));
#pragma unroll
    for (int off = 16; off >= 1; off >>= 1)
        m = fmaxf(m, __shfl_xor(m, off, 32));
    if (lane == 0) scores[row] = m;
}

// ---------------------------------------------------------------------------
// Kernel 2: per-batch radix-select of the exact 1500th-largest score, then a
// single-wave ballot compaction of the top-1500 set (ties: lowest index first,
// matching jax top_k). For each selected box, store packed [lo|hi<<16] row and
// col intervals, replicating the reference's scale ordering exactly:
//   lx = floor(x1*valid_h), ly = floor(y1*valid_w),
//   rx = floor(x2*valid_h), ry = floor(y2*valid_w).
// ---------------------------------------------------------------------------
__global__ __launch_bounds__(1024) void select_kernel(const float* __restrict__ coord,
                                                      const float* __restrict__ scores,
                                                      const int* __restrict__ vfs,
                                                      unsigned* __restrict__ rowpack,
                                                      unsigned* __restrict__ colpack) {
    int b = blockIdx.x;
    const float* s = scores + (size_t)b * NQ;

    __shared__ unsigned hist[256];
    __shared__ unsigned sh_prefix;
    __shared__ unsigned sh_kremain;

    unsigned t = threadIdx.x;
    if (t == 0) { sh_prefix = 0u; sh_kremain = TOPK; }
    __syncthreads();

    for (int pass = 0; pass < 4; ++pass) {
        int shift = 24 - 8 * pass;
        if (t < 256) hist[t] = 0u;
        __syncthreads();
        unsigned pfx = sh_prefix;
        for (int j = (int)t; j < NQ; j += 1024) {
            unsigned bits = __float_as_uint(s[j]);
            bool match = (pass == 0) || ((bits >> (shift + 8)) == (pfx >> (shift + 8)));
            if (match) atomicAdd(&hist[(bits >> shift) & 255u], 1u);
        }
        __syncthreads();
        if (t == 0) {
            unsigned kr = sh_kremain;
            for (int v = 255; v >= 0; --v) {
                unsigned c = hist[v];
                if (c >= kr) { sh_prefix = pfx | ((unsigned)v << shift); break; }
                kr -= c;
            }
            sh_kremain = kr;
        }
        __syncthreads();
    }

    unsigned T         = sh_prefix;    // exact 1500th-largest value (bit pattern)
    unsigned needed_eq = sh_kremain;   // # of ==T elements to take (lowest index)
    float vh = (float)vfs[b * 2 + 0];
    float vw = (float)vfs[b * 2 + 1];

    if (t < 32) {
        unsigned lane  = t;
        unsigned below = (1u << lane) - 1u;
        unsigned out_base = 0u, eq_base = 0u;
        for (int j0 = 0; j0 < NQ; j0 += 32) {
            int j = j0 + (int)lane;
            unsigned bits = (j < NQ) ? __float_as_uint(s[j]) : 0u;
            bool gt = (j < NQ) && (bits > T);
            bool eq = (j < NQ) && (bits == T);
            unsigned meq = (unsigned)__ballot(eq);
            bool take = gt || (eq && (eq_base + (unsigned)__popc(meq & below) < needed_eq));
            unsigned mtk = (unsigned)__ballot(take);
            if (take) {
                unsigned pos = out_base + (unsigned)__popc(mtk & below);
                const float* c4 = coord + ((size_t)b * NQ + j) * 4;
                float cx = c4[0], cy = c4[1], w = c4[2], h = c4[3];
                float x1 = cx - 0.5f * w, y1 = cy - 0.5f * h;
                float x2 = cx + 0.5f * w, y2 = cy + 0.5f * h;
                int lx = (int)floorf(x1 * vh);
                int ly = (int)floorf(y1 * vw);
                int rx = (int)floorf(x2 * vh);
                int ry = (int)floorf(y2 * vw);
                lx = lx < 0 ? 0 : (lx > 65535 ? 65535 : lx);
                ly = ly < 0 ? 0 : (ly > 65535 ? 65535 : ly);
                rx = rx < 0 ? 0 : (rx > 65535 ? 65535 : rx);
                ry = ry < 0 ? 0 : (ry > 65535 ? 65535 : ry);
                rowpack[(size_t)b * KPAD + pos] = (unsigned)ly | ((unsigned)ry << 16);
                colpack[(size_t)b * KPAD + pos] = (unsigned)lx | ((unsigned)rx << 16);
            }
            out_base += (unsigned)__popc(mtk);
            eq_base  += (unsigned)__popc(meq);
        }
        for (unsigned p = out_base + lane; p < KPAD; p += 32) {
            rowpack[(size_t)b * KPAD + p] = 0u;
            colpack[(size_t)b * KPAD + p] = 0u;
        }
    }
}

// ---------------------------------------------------------------------------
// Kernel 3: covered = (R^T x C) > 0 via v_wmma_f32_16x16x32_f16.
// Strip decomposition: 7 waves per batch; wave mt owns the 16-row strip
// [16*mt, 16*mt+16) x full width (7 tiles, 7 f32 accumulators). Per K-chunk the
// A-fragment (rows) is built ONCE and reused by 7 back-to-back WMMAs against 7
// B-fragments (distinct D registers -> no D->A/B RAW hazard).
// Fragment lane layouts (wave32, 16-bit):
//   A 16x32: lane L -> M = L%16; VGPR v -> K = (v<4?0:16)+(L>=16?8:0)+(v%4)*2
//   B 32x16: lane L -> N = L%16; VGPR v -> K = (v<4?0:16)+(L>=16?8:0)+(v%4)*2
// Elements synthesized from packed [lo|hi<<16] intervals in LDS (broadcast
// ds_load_b64, conflict-free). Epilogue fuses padding + -1e20 select using the
// 16x16 f32 C/D layout: lane L -> N = L%16, VGPR r -> M = r + 8*(L>=16).
// ---------------------------------------------------------------------------
__global__ __launch_bounds__(224) void mask_wmma_kernel(const unsigned* __restrict__ rowpack,
                                                        const unsigned* __restrict__ colpack,
                                                        const int* __restrict__ vfs,
                                                        float* __restrict__ out) {
    int b = blockIdx.x;
    __shared__ unsigned srow[KPAD];
    __shared__ unsigned scol[KPAD];
    for (int i = threadIdx.x; i < KPAD; i += 224) {
        srow[i] = rowpack[(size_t)b * KPAD + i];
        scol[i] = colpack[(size_t)b * KPAD + i];
    }
    __syncthreads();

    int mt    = threadIdx.x >> 5;      // 7 waves -> 7 row strips
    int lane  = threadIdx.x & 31;
    int hl    = lane & 15;
    int khalf = (lane >> 4) & 1;
    int h     = mt * 16 + hl;          // A-matrix row this lane represents

    v8f acc[7];
#pragma unroll
    for (int nt = 0; nt < 7; ++nt) acc[nt] = (v8f){};

#pragma unroll 1
    for (int kc = 0; kc < KPAD / 32; ++kc) {          // 47 chunks of K=32
        // Build A-fragment once per chunk (shared by all 7 column tiles).
        v16h afrag;
        int kA = kc * 32 + khalf * 8;
#pragma unroll
        for (int v = 0; v < 8; ++v) {
            int ka = kA + ((v & 4) << 2) + ((v & 3) << 1);
            unsigned r0 = srow[ka], r1 = srow[ka + 1];
            afrag[2 * v]     = (h >= (int)(r0 & 0xffffu) && h < (int)(r0 >> 16))
                                   ? (_Float16)1.0f : (_Float16)0.0f;
            afrag[2 * v + 1] = (h >= (int)(r1 & 0xffffu) && h < (int)(r1 >> 16))
                                   ? (_Float16)1.0f : (_Float16)0.0f;
        }
        int kB = kc * 32 + khalf * 8;
#pragma unroll
        for (int nt = 0; nt < 7; ++nt) {
            int w = nt * 16 + hl;
            v16h bfrag;
#pragma unroll
            for (int v = 0; v < 8; ++v) {
                int kb = kB + ((v & 4) << 2) + ((v & 3) << 1);
                unsigned c0 = scol[kb], c1 = scol[kb + 1];
                bfrag[2 * v]     = (w >= (int)(c0 & 0xffffu) && w < (int)(c0 >> 16))
                                       ? (_Float16)1.0f : (_Float16)0.0f;
                bfrag[2 * v + 1] = (w >= (int)(c1 & 0xffffu) && w < (int)(c1 >> 16))
                                       ? (_Float16)1.0f : (_Float16)0.0f;
            }
            acc[nt] = __builtin_amdgcn_wmma_f32_16x16x32_f16(
                /*neg_a=*/false, afrag, /*neg_b=*/false, bfrag,
                /*c_mod=*/(short)0, acc[nt], /*reuse_a=*/false, /*reuse_b=*/false);
        }
    }

    int vvh = vfs[b * 2 + 0];
    int vvw = vfs[b * 2 + 1];
#pragma unroll
    for (int nt = 0; nt < 7; ++nt) {
#pragma unroll
        for (int r = 0; r < 8; ++r) {
            int hh = mt * 16 + r + khalf * 8;
            int ww = nt * 16 + hl;
            if (hh < FH && ww < FW) {
                bool covered = acc[nt][r] > 0.0f;
                bool pad = (hh >= vvh) || (ww >= vvw);
                out[(size_t)b * (FH * FW) + hh * FW + ww] =
                    (covered && !pad) ? 0.0f : -1.0e20f;
            }
        }
    }
}

// ---------------------------------------------------------------------------
extern "C" void kernel_launch(void* const* d_in, const int* in_sizes, int n_in,
                              void* d_out, int out_size, void* d_ws, size_t ws_size,
                              hipStream_t stream) {
    const float* coord = (const float*)d_in[0];   // (64,3000,4) f32
    const float* cls   = (const float*)d_in[1];   // (64,3000,256) f32
    const int*   vfs   = (const int*)d_in[2];     // (64,2) i32
    float*       out   = (float*)d_out;           // (64,10000) f32

    char* ws = (char*)d_ws;
    float*    scores  = (float*)ws;                               // 192000 f32
    unsigned* rowpack = (unsigned*)(ws + (size_t)BS * NQ * 4);    // 64*1504 u32
    unsigned* colpack = rowpack + (size_t)BS * KPAD;              // 64*1504 u32

    scores_kernel<<<(BS * NQ) / 8, 256, 0, stream>>>(cls, scores);
    select_kernel<<<BS, 1024, 0, stream>>>(coord, scores, vfs, rowpack, colpack);
    mask_wmma_kernel<<<BS, 224, 0, stream>>>(rowpack, colpack, vfs, out);
}
